// GATLayer_3496103379257
// MI455X (gfx1250) — compile-verified, matrix operand order checked by reading, not run on hardware
//
#include <hip/hip_runtime.h>
#include <hip/hip_bf16.h>

// GATv2 layer for MI455X (gfx1250, wave32, WMMA).
// GEMMs use v_wmma_f32_16x16x32_bf16 (bf16 in / f32 accum); edge phase is
// wave-per-edge with half-wave-per-head reductions and L2 float atomics.

typedef __attribute__((ext_vector_type(16))) __bf16 v16bf;
typedef __attribute__((ext_vector_type(8)))  float  v8f;

#define NEG_SLOPE 0.2f
#define NEG_FLT_MAX (-3.402823466e38f)

__device__ __forceinline__ __bf16 f2bf(float f) {
  // round-to-nearest-even fp32 -> bf16
  union { unsigned short u; __bf16 b; } cvt;
  unsigned u32 = __float_as_uint(f);
  cvt.u = (unsigned short)((u32 + 0x7FFFu + ((u32 >> 16) & 1u)) >> 16);
  return cvt.b;
}

// ---------------- prep kernels ----------------

__global__ void k_cvt_bf16(const float* __restrict__ src, __bf16* __restrict__ dst, int n) {
  int i = blockIdx.x * blockDim.x + threadIdx.x;
  int stride = gridDim.x * blockDim.x;
  for (; i < n; i += stride) dst[i] = f2bf(src[i]);
}

__global__ void k_fill(float* __restrict__ p, float v, int n) {
  int i = blockIdx.x * blockDim.x + threadIdx.x;
  if (i < n) p[i] = v;
}

__global__ void k_init_agg(float* __restrict__ agg, const float* __restrict__ bias, int total) {
  int i = blockIdx.x * blockDim.x + threadIdx.x;
  if (i < total) agg[i] = bias[i & 255];   // HD == 256
}

// ---------------- WMMA GEMM: C[M,O] = A[M,K] * W[O,K]^T + bias ----------------
// One wave per 16x16 C tile. A,W bf16 row-major. M % 16 == 0, K % 32 == 0, O % 16 == 0.

__global__ void k_gemm_bf16(const __bf16* __restrict__ A, const __bf16* __restrict__ W,
                            const float* __restrict__ bias, float* __restrict__ C,
                            int M, int K, int O) {
  const int wave = threadIdx.x >> 5;
  const int lane = threadIdx.x & 31;
  const int otiles = O >> 4;
  const int mtiles = M >> 4;
  const int tid = blockIdx.x * (blockDim.x >> 5) + wave;
  if (tid >= mtiles * otiles) return;            // wave-uniform exit (EXEC all-ones for WMMA)
  const int mt = tid / otiles, ot = tid - mt * otiles;
  const int m0 = mt << 4, o0 = ot << 4;
  const int lh = lane & 15, hi = lane >> 4;

  const __bf16* arow = A + (size_t)(m0 + lh) * K + hi * 8;
  const __bf16* brow = W + (size_t)(o0 + lh) * K + hi * 8;

  v8f acc = {};
  for (int kk = 0; kk < K; kk += 32) {
    v16bf a, b;
    ((uint4*)&a)[0] = *(const uint4*)(arow + kk);
    ((uint4*)&a)[1] = *(const uint4*)(arow + kk + 16);
    ((uint4*)&b)[0] = *(const uint4*)(brow + kk);
    ((uint4*)&b)[1] = *(const uint4*)(brow + kk + 16);
    acc = __builtin_amdgcn_wmma_f32_16x16x32_bf16(false, a, false, b, (short)0, acc,
                                                  false, false);
  }
  const int col = o0 + lh;
  const float bv = bias[col];
  const int mrow = m0 + hi * 8;
#pragma unroll
  for (int r = 0; r < 8; ++r)
    C[(size_t)(mrow + r) * O + col] = acc[r] + bv;
}

// Same GEMM but A is fp32 (converted to bf16 in-register while loading).
__global__ void k_gemm_f32a(const float* __restrict__ A, const __bf16* __restrict__ W,
                            const float* __restrict__ bias, float* __restrict__ C,
                            int M, int K, int O) {
  const int wave = threadIdx.x >> 5;
  const int lane = threadIdx.x & 31;
  const int otiles = O >> 4;
  const int mtiles = M >> 4;
  const int tid = blockIdx.x * (blockDim.x >> 5) + wave;
  if (tid >= mtiles * otiles) return;
  const int mt = tid / otiles, ot = tid - mt * otiles;
  const int m0 = mt << 4, o0 = ot << 4;
  const int lh = lane & 15, hi = lane >> 4;

  const float*  arow = A + (size_t)(m0 + lh) * K + hi * 8;
  const __bf16* brow = W + (size_t)(o0 + lh) * K + hi * 8;

  v8f acc = {};
  for (int kk = 0; kk < K; kk += 32) {
    float4 f0 = *(const float4*)(arow + kk);
    float4 f1 = *(const float4*)(arow + kk + 4);
    float4 f2 = *(const float4*)(arow + kk + 16);
    float4 f3 = *(const float4*)(arow + kk + 20);
    v16bf a, b;
    a[0]  = f2bf(f0.x); a[1]  = f2bf(f0.y); a[2]  = f2bf(f0.z); a[3]  = f2bf(f0.w);
    a[4]  = f2bf(f1.x); a[5]  = f2bf(f1.y); a[6]  = f2bf(f1.z); a[7]  = f2bf(f1.w);
    a[8]  = f2bf(f2.x); a[9]  = f2bf(f2.y); a[10] = f2bf(f2.z); a[11] = f2bf(f2.w);
    a[12] = f2bf(f3.x); a[13] = f2bf(f3.y); a[14] = f2bf(f3.z); a[15] = f2bf(f3.w);
    ((uint4*)&b)[0] = *(const uint4*)(brow + kk);
    ((uint4*)&b)[1] = *(const uint4*)(brow + kk + 16);
    acc = __builtin_amdgcn_wmma_f32_16x16x32_bf16(false, a, false, b, (short)0, acc,
                                                  false, false);
  }
  const int col = o0 + lh;
  const float bv = bias[col];
  const int mrow = m0 + hi * 8;
#pragma unroll
  for (int r = 0; r < 8; ++r)
    C[(size_t)(mrow + r) * O + col] = acc[r] + bv;
}

// ---------------- edge phase (wave per edge; lanes 0-15 head 0, 16-31 head 1) ----------------

__global__ void k_score(const float* __restrict__ xl, const float* __restrict__ xr,
                        const float* __restrict__ att, const int* __restrict__ ei,
                        int E, int Etot, float* __restrict__ score, float* __restrict__ mbuf) {
  const int gw = (blockIdx.x * blockDim.x + threadIdx.x) >> 5;
  if (gw >= Etot) return;
  const int lane = threadIdx.x & 31;
  const int h = lane >> 4, l = lane & 15;
  int src, dst;
  if (gw < E) { src = ei[gw]; dst = ei[E + gw]; } else { src = dst = gw - E; }

  const float* pl = xl + (size_t)src * 256 + h * 128;
  const float* pr = xr + (size_t)dst * 256 + h * 128;
  const float* pa = att + h * 128;
  float s = 0.f;
#pragma unroll
  for (int j = 0; j < 8; ++j) {
    const int d = l + j * 16;
    float v = pl[d] + pr[d];
    v = (v > 0.f) ? v : NEG_SLOPE * v;
    s += v * pa[d];
  }
  s += __shfl_xor(s, 1, 32);
  s += __shfl_xor(s, 2, 32);
  s += __shfl_xor(s, 4, 32);
  s += __shfl_xor(s, 8, 32);
  if (l == 0) {
    score[(size_t)gw * 2 + h] = s;
    float* addr = mbuf + (size_t)dst * 2 + h;      // float atomic max via int-bit trick
    if (s >= 0.f) atomicMax((int*)addr, __float_as_int(s));
    else          atomicMin((unsigned int*)addr, __float_as_uint(s));
  }
}

__global__ void k_expsum(const int* __restrict__ ei, int E, int Etot,
                         const float* __restrict__ mbuf, float* __restrict__ score,
                         float* __restrict__ denom) {
  const int e = blockIdx.x * blockDim.x + threadIdx.x;
  if (e >= Etot) return;
  const int dst = (e < E) ? ei[E + e] : (e - E);
#pragma unroll
  for (int h = 0; h < 2; ++h) {
    const float a = __expf(score[(size_t)e * 2 + h] - mbuf[(size_t)dst * 2 + h]);
    score[(size_t)e * 2 + h] = a;
    atomicAdd(&denom[(size_t)dst * 2 + h], a);
  }
}

__global__ void k_aggregate(const float* __restrict__ xl, const int* __restrict__ ei,
                            int E, int Etot, const float* __restrict__ score,
                            const float* __restrict__ denom, float* __restrict__ agg) {
  const int gw = (blockIdx.x * blockDim.x + threadIdx.x) >> 5;
  if (gw >= Etot) return;
  const int lane = threadIdx.x & 31;
  const int h = lane >> 4, l = lane & 15;
  int src, dst;
  if (gw < E) { src = ei[gw]; dst = ei[E + gw]; } else { src = dst = gw - E; }

  const float alpha = score[(size_t)gw * 2 + h] / (denom[(size_t)dst * 2 + h] + 1e-16f);
  const float* pl = xl + (size_t)src * 256 + h * 128;
  float* po = agg + (size_t)dst * 256 + h * 128;
#pragma unroll
  for (int j = 0; j < 8; ++j) {
    const int d = l + j * 16;
    atomicAdd(po + d, pl[d] * alpha);
  }
}

// ---------------- launcher ----------------

extern "C" void kernel_launch(void* const* d_in, const int* in_sizes, int n_in,
                              void* d_out, int out_size, void* d_ws, size_t ws_size,
                              hipStream_t stream) {
  (void)n_in; (void)out_size; (void)ws_size;

  const float* x   = (const float*)d_in[0];
  const int*   ei  = (const int*)d_in[1];
  const float* Wl  = (const float*)d_in[2];
  const float* bl  = (const float*)d_in[3];
  const float* Wr  = (const float*)d_in[4];
  const float* br  = (const float*)d_in[5];
  const float* att = (const float*)d_in[6];
  const float* bc  = (const float*)d_in[7];
  const float* Wo  = (const float*)d_in[8];
  const float* bo  = (const float*)d_in[9];
  float* out = (float*)d_out;

  const int D = 128, H = 2, HD = 256;
  const int N    = in_sizes[0] / D;
  const int E    = in_sizes[1] / 2;
  const int Etot = E + N;

  // workspace carve-out (256B aligned)
  char* ws = (char*)d_ws;
  size_t off = 0;
  auto carve = [&](size_t bytes) -> void* {
    void* p = ws + off;
    off = (off + bytes + 255) & ~(size_t)255;
    return p;
  };
  __bf16* xbf   = (__bf16*)carve((size_t)N * D * 2);
  __bf16* Wlbf  = (__bf16*)carve((size_t)HD * D * 2);
  __bf16* Wrbf  = (__bf16*)carve((size_t)HD * D * 2);
  __bf16* Wobf  = (__bf16*)carve((size_t)D * HD * 2);
  float*  xl    = (float*)carve((size_t)N * HD * 4);
  float*  xr    = (float*)carve((size_t)N * HD * 4);
  float*  agg   = (float*)carve((size_t)N * HD * 4);
  float*  score = (float*)carve((size_t)Etot * H * 4);
  float*  mbuf  = (float*)carve((size_t)N * H * 4);
  float*  denom = (float*)carve((size_t)N * H * 4);

  const int THR = 256;

  // precision conversion + init
  k_cvt_bf16<<<(N * D + THR - 1) / THR, THR, 0, stream>>>(x,  xbf,  N * D);
  k_cvt_bf16<<<(HD * D + THR - 1) / THR, THR, 0, stream>>>(Wl, Wlbf, HD * D);
  k_cvt_bf16<<<(HD * D + THR - 1) / THR, THR, 0, stream>>>(Wr, Wrbf, HD * D);
  k_cvt_bf16<<<(D * HD + THR - 1) / THR, THR, 0, stream>>>(Wo, Wobf, D * HD);
  k_fill<<<(N * H + THR - 1) / THR, THR, 0, stream>>>(mbuf, NEG_FLT_MAX, N * H);
  k_fill<<<(N * H + THR - 1) / THR, THR, 0, stream>>>(denom, 0.f, N * H);
  k_init_agg<<<(N * HD + THR - 1) / THR, THR, 0, stream>>>(agg, bc, N * HD);

  // projections: xl = x@Wl^T + bl ; xr = x@Wr^T + br   (WMMA bf16)
  const int tilesP  = (N >> 4) * (HD >> 4);
  const int blocksP = (tilesP + 3) / 4;                 // 4 waves / 128-thread block
  k_gemm_bf16<<<blocksP, 128, 0, stream>>>(xbf, Wlbf, bl, xl, N, D, HD);
  k_gemm_bf16<<<blocksP, 128, 0, stream>>>(xbf, Wrbf, br, xr, N, D, HD);

  // attention: score -> segment max -> exp/segment sum -> weighted scatter
  const int blocksW = (Etot * 32 + THR - 1) / THR;      // wave per edge
  k_score<<<blocksW, THR, 0, stream>>>(xl, xr, att, ei, E, Etot, score, mbuf);
  k_expsum<<<(Etot + THR - 1) / THR, THR, 0, stream>>>(ei, E, Etot, mbuf, score, denom);
  k_aggregate<<<blocksW, THR, 0, stream>>>(xl, ei, E, Etot, score, denom, agg);

  // final projection: out = agg@Wo^T + bo   (WMMA, fp32 A converted in-register)
  const int tilesO = (N >> 4) * (D >> 4);
  k_gemm_f32a<<<(tilesO + 3) / 4, 128, 0, stream>>>(agg, Wobf, bo, out, N, HD, D);
}
